// SingleLayerLSTM_87823491268730
// MI455X (gfx1250) — compile-verified, matrix operand order checked by reading, not run on hardware
//
#include <hip/hip_runtime.h>
#include <math.h>

// Problem dims
#define BB    128      // batch
#define TT    512      // time
#define IIN   128      // input feature
#define HH    1024     // hidden
#define NGATE 4096     // 3H (fiz) + H (r) concatenated gate columns
#define KAD   1152     // IN + H (K dim of gate GEMM)
#define ONN   64       // output feature

// Persistent grid: 64 blocks x 8 wave32 = 512 waves = exactly one 32x32
// gate macro-tile per wave per timestep.
#define NBLK  64
#define NTHR  256
#define NWAVE (NBLK * 8)

typedef __attribute__((ext_vector_type(16))) __bf16          v16bf;
typedef __attribute__((ext_vector_type(8)))  float           v8f;
typedef __attribute__((ext_vector_type(8)))  unsigned short  ushort8;
typedef __attribute__((ext_vector_type(16))) unsigned short  ushort16;

// ---------------- workspace layout (bytes, 256-aligned) ----------------
#define WCAT_OFF   0u                                   // bf16 WcatT[NGATE][KAD]    9,437,184 B
#define WOUT_OFF   9437184u                             // bf16 WoutT[ONN][HH]         131,072 B
#define ABUF_OFF   9568256u                             // bf16 Abuf[BB][KAD]          294,912 B
#define GATE_OFF   9863168u                             // f32  gates[BB][NGATE]     2,097,152 B
#define CST_OFF    11960320u                            // f32  c[BB][HH]              524,288 B
#define BAR_OFF    12484608u                            // 2x u32 barrier state
#define WS_NEEDED  12484864u

// ---------------- helpers ----------------
__device__ __forceinline__ unsigned short f2bf(float x) {
  unsigned u = __builtin_bit_cast(unsigned, x);
  unsigned r = u + 0x7FFFu + ((u >> 16) & 1u);   // round-to-nearest-even
  return (unsigned short)(r >> 16);
}

__device__ __forceinline__ float sigmoid_f(float x) {
  return 1.0f / (1.0f + __expf(-x));
}
__device__ __forceinline__ float tanh_f(float x) {
  float a = fabsf(x);
  float e = __expf(-2.0f * a);
  float t = (1.0f - e) / (1.0f + e);
  return copysignf(t, x);
}

// Load a 16-bit A/B fragment (16x32 tile, this lane's 16 bf16 values).
// ISA layout: lane<16 -> row=lane, K = {k0..k0+7, k0+16..k0+23}
//             lane>=16 -> row=lane-16, K = {k0+8..k0+15, k0+24..k0+31}
// Caller pre-biases the pointer by row*stride + (lane>>4)*8.
__device__ __forceinline__ v16bf load_frag16(const unsigned short* p) {
  ushort8 lo = *(const ushort8*)(p);
  ushort8 hi = *(const ushort8*)(p + 16);
  ushort16 w = __builtin_shufflevector(lo, hi, 0, 1, 2, 3, 4, 5, 6, 7,
                                       8, 9, 10, 11, 12, 13, 14, 15);
  return __builtin_bit_cast(v16bf, w);
}

__device__ __forceinline__ v8f wmma_bf16(v16bf a, v16bf b, v8f c) {
  return __builtin_amdgcn_wmma_f32_16x16x32_bf16(
      false, a, false, b, (short)0, c, false, false);
}

// Device-wide barrier: arrival counter + monotonically increasing release
// generation. Counters are zeroed by the prep kernel each launch.
__device__ __forceinline__ void grid_barrier(unsigned* cnt, unsigned* rel,
                                             unsigned gen) {
  __syncthreads();
  if (threadIdx.x == 0) {
    __threadfence();
    unsigned prev = __hip_atomic_fetch_add(cnt, 1u, __ATOMIC_ACQ_REL,
                                           __HIP_MEMORY_SCOPE_AGENT);
    if (prev == gen * NBLK + (NBLK - 1)) {
      __hip_atomic_store(rel, gen + 1u, __ATOMIC_RELEASE,
                         __HIP_MEMORY_SCOPE_AGENT);
    } else {
      while (__hip_atomic_load(rel, __ATOMIC_ACQUIRE,
                               __HIP_MEMORY_SCOPE_AGENT) <= gen) {
        __builtin_amdgcn_s_sleep(1);
      }
    }
    __threadfence();
  }
  __syncthreads();
}

// ---------------- prep kernels ----------------
// WcatT[n][k] = (n < 3H) ? kernel_fiz[k][n] : kernel_r[k][n-3H]   (bf16)
// WoutT[o][k] = W_out[k][o]                                        (bf16)
__global__ void lstm_prep_weights(const float* __restrict__ kfiz,
                                  const float* __restrict__ kr,
                                  const float* __restrict__ wout,
                                  unsigned short* __restrict__ WcatT,
                                  unsigned short* __restrict__ WoutT) {
  const int total = NGATE * KAD + ONN * HH;
  for (int idx = blockIdx.x * blockDim.x + threadIdx.x; idx < total;
       idx += gridDim.x * blockDim.x) {
    if (idx < NGATE * KAD) {
      int n = idx / KAD;
      int k = idx - n * KAD;
      float v = (n < 3 * HH) ? kfiz[k * (3 * HH) + n]
                             : kr[k * HH + (n - 3 * HH)];
      WcatT[idx] = f2bf(v);
    } else {
      int r = idx - NGATE * KAD;
      int o = r / HH;
      int k = r - o * HH;
      WoutT[r] = f2bf(wout[k * ONN + o]);
    }
  }
}

// Abuf[b][0:IN) = bf16(u[b,0,:]); Abuf[b][IN:IN+H) = bf16(h0); c = c0.
// Also zero the barrier counters (deterministic per launch).
__global__ void lstm_prep_state(const float* __restrict__ u,
                                const float* __restrict__ x0,
                                unsigned short* __restrict__ Abuf,
                                float* __restrict__ cst,
                                unsigned* __restrict__ bar) {
  const int total = BB * KAD + BB * HH;
  int gid = blockIdx.x * blockDim.x + threadIdx.x;
  if (gid == 0) { bar[0] = 0u; bar[1] = 0u; }
  for (int idx = gid; idx < total; idx += gridDim.x * blockDim.x) {
    if (idx < BB * KAD) {
      int b = idx / KAD;
      int k = idx - b * KAD;
      float v = (k < IIN) ? u[b * TT * IIN + k]            // u[b,0,k]
                          : x0[b * 2 * HH + (k - IIN)];    // h0
      Abuf[idx] = f2bf(v);
    } else {
      int r = idx - BB * KAD;
      int b = r / HH;
      int j = r - b * HH;
      cst[r] = x0[b * 2 * HH + HH + j];                    // c0
    }
  }
}

// ---------------- persistent recurrent kernel ----------------
__global__ __launch_bounds__(NTHR) void lstm_persist(
    const float* __restrict__ u, const float* __restrict__ bias_fiz,
    const float* __restrict__ bias_r, const float* __restrict__ b_out,
    const unsigned short* __restrict__ WcatT,
    const unsigned short* __restrict__ WoutT,
    unsigned short* __restrict__ Abuf, float* __restrict__ gates,
    float* __restrict__ cst, float* __restrict__ y, unsigned* bar) {
  const int tid   = threadIdx.x;
  const int lane  = tid & 31;
  const int wave  = tid >> 5;
  const int gwave = blockIdx.x * 8 + wave;

  const int fragRow  = lane & 15;          // row (A) / col (B^T) within tile
  const int fragKoff = (lane >> 4) * 8;    // K sub-offset for this half-wave
  const int accRow   = (lane >> 4) * 8;    // C/D: lanes>=16 hold M+8
  const int accCol   = lane & 15;

  unsigned gen = 0;

  for (int t = 0; t < TT; ++t) {
    // ---- Stage A: 32x32 register-blocked gate GEMM for step t,
    //      plus y(t-1) projection tiles (waves 0..31, second pass) ----
    const int ntiles = 512 + ((t > 0) ? 32 : 0);
    for (int tile = gwave; tile < ntiles; tile += NWAVE) {
      if (tile < 512) {
        const int mm = tile >> 7;     // 0..3   (32-row macro tile)
        const int nn = tile & 127;    // 0..127 (32-col macro tile)
        const unsigned short* ap0 =
            Abuf + (mm * 32 + fragRow) * KAD + fragKoff;
        const unsigned short* ap1 = ap0 + 16 * KAD;
        const unsigned short* bp0 =
            WcatT + (nn * 32 + fragRow) * KAD + fragKoff;
        const unsigned short* bp1 = bp0 + 16 * KAD;
        v8f a00 = {}, a01 = {}, a10 = {}, a11 = {};
#pragma unroll 2
        for (int k = 0; k < KAD; k += 32) {
          v16bf fa0 = load_frag16(ap0 + k);
          v16bf fa1 = load_frag16(ap1 + k);
          v16bf fb0 = load_frag16(bp0 + k);
          v16bf fb1 = load_frag16(bp1 + k);
          a00 = wmma_bf16(fa0, fb0, a00);
          a01 = wmma_bf16(fa0, fb1, a01);
          a10 = wmma_bf16(fa1, fb0, a10);
          a11 = wmma_bf16(fa1, fb1, a11);
        }
        const int gm = mm * 32 + accRow;
        const int gn = nn * 32 + accCol;
#pragma unroll
        for (int j = 0; j < 8; ++j) {
          gates[(gm + j) * NGATE + gn]           = a00[j];
          gates[(gm + j) * NGATE + gn + 16]      = a01[j];
          gates[(gm + 16 + j) * NGATE + gn]      = a10[j];
          gates[(gm + 16 + j) * NGATE + gn + 16] = a11[j];
        }
      } else {
        const int ot = tile - 512;    // 0..31
        const int mt = ot >> 2;       // 0..7
        const int nt = ot & 3;        // 0..3
        const unsigned short* ap =
            Abuf + (mt * 16 + fragRow) * KAD + IIN + fragKoff;  // h region
        const unsigned short* bp =
            WoutT + (nt * 16 + fragRow) * HH + fragKoff;
        v8f acc = {};
#pragma unroll 4
        for (int k = 0; k < HH; k += 32) {
          acc = wmma_bf16(load_frag16(ap + k), load_frag16(bp + k), acc);
        }
        const int gm = mt * 16 + accRow;        // batch row
        const int gn = nt * 16 + accCol;        // output col
        const float bo = b_out[gn];
#pragma unroll
        for (int j = 0; j < 8; ++j)
          y[((gm + j) * TT + (t - 1)) * ONN + gn] = acc[j] + bo;
      }
    }
    grid_barrier(bar, bar + 1, gen++);

    // ---- Stage B: activations + state update + stage next A-matrix ----
    for (int idx = blockIdx.x * NTHR + tid; idx < BB * HH;
         idx += NBLK * NTHR) {
      const int b = idx >> 10;
      const int j = idx & 1023;
      const float* g = gates + b * NGATE;
      float f  = sigmoid_f(g[j]            + bias_fiz[j]);
      float i_ = sigmoid_f(g[HH + j]       + bias_fiz[HH + j]);
      float z  = sigmoid_f(g[2 * HH + j]   + bias_fiz[2 * HH + j]);
      float r  = tanh_f   (g[3 * HH + j]   + bias_r[j]);
      float c  = f * cst[idx] + i_ * r;
      cst[idx] = c;
      float h  = z * tanh_f(c);
      Abuf[b * KAD + IIN + j] = f2bf(h);
    }
    if (t + 1 < TT) {
      for (int idx = blockIdx.x * NTHR + tid; idx < BB * IIN;
           idx += NBLK * NTHR) {
        const int b = idx >> 7;
        const int i2 = idx & 127;
        Abuf[b * KAD + i2] = f2bf(u[(b * TT + (t + 1)) * IIN + i2]);
      }
    }
    grid_barrier(bar, bar + 1, gen++);
  }

  // ---- Epilogue: y(T-1) ----
  for (int tile = gwave; tile < 32; tile += NWAVE) {
    const int mt = tile >> 2;
    const int nt = tile & 3;
    const unsigned short* ap =
        Abuf + (mt * 16 + fragRow) * KAD + IIN + fragKoff;
    const unsigned short* bp =
        WoutT + (nt * 16 + fragRow) * HH + fragKoff;
    v8f acc = {};
#pragma unroll 4
    for (int k = 0; k < HH; k += 32) {
      acc = wmma_bf16(load_frag16(ap + k), load_frag16(bp + k), acc);
    }
    const int gm = mt * 16 + accRow;
    const int gn = nt * 16 + accCol;
    const float bo = b_out[gn];
#pragma unroll
    for (int j = 0; j < 8; ++j)
      y[((gm + j) * TT + (TT - 1)) * ONN + gn] = acc[j] + bo;
  }
}

// ---------------- host launch ----------------
extern "C" void kernel_launch(void* const* d_in, const int* in_sizes, int n_in,
                              void* d_out, int out_size, void* d_ws,
                              size_t ws_size, hipStream_t stream) {
  (void)in_sizes; (void)n_in; (void)out_size; (void)ws_size;
  const float* u        = (const float*)d_in[0];
  const float* x0       = (const float*)d_in[1];
  const float* kfiz     = (const float*)d_in[2];
  const float* bias_fiz = (const float*)d_in[3];
  const float* kr       = (const float*)d_in[4];
  const float* bias_r   = (const float*)d_in[5];
  const float* wout     = (const float*)d_in[6];
  const float* b_out    = (const float*)d_in[7];
  float* y = (float*)d_out;

  char* ws = (char*)d_ws;
  unsigned short* WcatT = (unsigned short*)(ws + WCAT_OFF);
  unsigned short* WoutT = (unsigned short*)(ws + WOUT_OFF);
  unsigned short* Abuf  = (unsigned short*)(ws + ABUF_OFF);
  float*          gates = (float*)(ws + GATE_OFF);
  float*          cst   = (float*)(ws + CST_OFF);
  unsigned*       bar   = (unsigned*)(ws + BAR_OFF);

  lstm_prep_weights<<<4096, 256, 0, stream>>>(kfiz, kr, wout, WcatT, WoutT);
  lstm_prep_state<<<1088, 256, 0, stream>>>(u, x0, Abuf, cst, bar);
  lstm_persist<<<NBLK, NTHR, 0, stream>>>(u, bias_fiz, bias_r, b_out, WcatT,
                                          WoutT, Abuf, gates, cst, y, bar);
}
